// STGNN_25881472926343
// MI455X (gfx1250) — matrix operand order repaired and checked by reading, not compiled, not run on hardware
//
#include <hip/hip_runtime.h>

// ---------------- problem constants (match reference) ----------------
#define N_NODES   20000
#define N_EDGES   640000
#define F_IN      128
#define N_GRAPHS  64
#define N_CLASSES 2

typedef __attribute__((ext_vector_type(16))) __bf16 v16bf;
typedef __attribute__((ext_vector_type(8)))  float  v8f;

// ---------------- helpers ----------------
__device__ __forceinline__ unsigned short f2bf(float f) {
    unsigned u = __builtin_bit_cast(unsigned, f);
    u += 0x7FFFu + ((u >> 16) & 1u);   // round to nearest even
    return (unsigned short)(u >> 16);
}
__device__ __forceinline__ float sigm(float x) { return 1.0f / (1.0f + __expf(-x)); }

// ---------------- small utility kernels ----------------
__global__ void k_fill_f32(float* p, float v, int n) {
    int i = blockIdx.x * blockDim.x + threadIdx.x;
    if (i < n) p[i] = v;
}

__global__ void k_add_vec(const float* a, const float* b, float* o, int n) {
    int i = blockIdx.x * blockDim.x + threadIdx.x;
    if (i < n) o[i] = a[i] + b[i];
}

// f32 -> bf16 (optionally fused ReLU)
__global__ void k_f32_to_bf16(const float* __restrict__ in, unsigned short* __restrict__ out,
                              int n, int relu) {
    int i = blockIdx.x * blockDim.x + threadIdx.x;
    if (i >= n) return;
    float v = in[i];
    if (relu) v = fmaxf(v, 0.0f);
    out[i] = f2bf(v);
}

// Pack weight into bf16, B-layout = [Nout][K] row-major.
// transpose==1: W given as math (K,Nout) row-major  -> Bp[n*K+k] = W[k*Nout+n]
// transpose==0: W given as (Nout,K) row-major (".T" usage) -> Bp[n*K+k] = W[n*K+k]
__global__ void k_pack_w(const float* __restrict__ W, unsigned short* __restrict__ Bp,
                         int K, int Nout, int transpose) {
    int idx = blockIdx.x * blockDim.x + threadIdx.x;
    if (idx >= K * Nout) return;
    int n = idx / K;
    int k = idx - n * K;
    float v = transpose ? W[(size_t)k * Nout + n] : W[(size_t)n * K + k];
    Bp[idx] = f2bf(v);
}

// ---------------- degree / norm ----------------
__global__ void k_deg(const int* __restrict__ dst, float* deg, int E) {
    int i = blockIdx.x * blockDim.x + threadIdx.x;
    if (i < E) atomicAdd(&deg[dst[i]], 1.0f);
}
__global__ void k_dinv(const float* __restrict__ deg, float* __restrict__ dinv, int n) {
    int i = blockIdx.x * blockDim.x + threadIdx.x;
    if (i < n) dinv[i] = rsqrtf(fmaxf(deg[i], 1.0f));   // deg >= 1 (self loop)
}

// ---------------- WMMA bf16 GEMM:  out[M,Nout] = A[M,K](bf16) * Bp^T + bias ----------------
// A row-major [M][K] bf16; Bp is [Nout][K] bf16 (i.e. math B transposed).
// One 16x16 tile per wave, 4 waves per block, K consumed 32 at a time via
// v_wmma_f32_16x16x32_bf16.  Fragment layouts per CDNA5 ISA 7.12.2:
//   A : lane L -> row M=L%16 ; elems 0..7 -> K=half*8+e ; elems 8..15 -> K=16+half*8+(e-8)
//   B : lane L -> col N=L%16 ; elems 0..7 -> K=half*8+e ; elems 8..15 -> K=16+half*8+(e-8)
//   C/D: lane L -> col N=L%16 ; vgpr v   -> row M=v+8*half
__global__ __launch_bounds__(128)
void k_gemm_bf16(const unsigned short* __restrict__ A, const unsigned short* __restrict__ Bp,
                 const float* __restrict__ bias, float* __restrict__ outf,
                 unsigned short* __restrict__ outbf, int M, int Nout, int K) {
    int wave = threadIdx.x >> 5;
    int lane = threadIdx.x & 31;
    int tiles_n = Nout >> 4;
    int tiles_total = (M >> 4) * tiles_n;
    int tile = blockIdx.x * 4 + wave;
    if (tile >= tiles_total) return;           // wave-uniform: EXEC stays all-ones
    int tm = tile / tiles_n;
    int tn = tile - tm * tiles_n;
    int m0 = tm << 4, n0 = tn << 4;
    int half = lane >> 4, row = lane & 15;

    v8f c = {0.f, 0.f, 0.f, 0.f, 0.f, 0.f, 0.f, 0.f};
    const uint4* Arow = reinterpret_cast<const uint4*>(A + (size_t)(m0 + row) * K); // 8 bf16 / uint4
    const uint4* Brow = reinterpret_cast<const uint4*>(Bp + (size_t)(n0 + row) * K);

    struct P { uint4 lo, hi; };
    int kchunks = K >> 5;
    for (int kb = 0; kb < kchunks; ++kb) {
        uint4 a0 = Arow[kb * 4 + half];          // K = kb*32 + half*8      (elems 0..7)
        uint4 a1 = Arow[kb * 4 + 2 + half];      // K = kb*32 + 16 + half*8 (elems 8..15)
        uint4 b0 = Brow[kb * 4 + half];          // K = kb*32 + half*8      (elems 0..7)
        uint4 b1 = Brow[kb * 4 + 2 + half];      // K = kb*32 + 16 + half*8 (elems 8..15)
        P pa{a0, a1}, pb{b0, b1};
        v16bf av = __builtin_bit_cast(v16bf, pa);
        v16bf bv = __builtin_bit_cast(v16bf, pb);
        c = __builtin_amdgcn_wmma_f32_16x16x32_bf16(false, av, false, bv,
                                                    (short)0, c, false, false);
    }

    float badd = bias ? bias[n0 + row] : 0.0f;
    int col = n0 + row;
    #pragma unroll
    for (int v = 0; v < 8; ++v) {
        int m = m0 + v + 8 * half;
        float val = c[v] + badd;
        size_t o = (size_t)m * Nout + col;
        if (outf)  outf[o]  = val;
        if (outbf) outbf[o] = f2bf(val);
    }
}

// ---------------- GCN aggregation ----------------
// acc[i][f] = dinv[i]^2 * h[i][f] + bias[f]   (self-loop message + bias)
__global__ void k_gcn_init(const float* __restrict__ h, const float* __restrict__ dinv,
                           const float* __restrict__ bias, float* __restrict__ acc,
                           int fshift, int total) {
    int i = blockIdx.x * blockDim.x + threadIdx.x;
    if (i >= total) return;
    int node = i >> fshift;
    int f = i & ((1 << fshift) - 1);
    float dv = dinv[node];
    acc[i] = dv * dv * h[i] + bias[f];
}

// acc[dst] += dinv[src]*dinv[dst] * h[src]   (float4 per thread)
__global__ void k_gcn_scatter(const int* __restrict__ src, const int* __restrict__ dst,
                              const float* __restrict__ dinv, const float* __restrict__ h,
                              float* __restrict__ acc, int gshift, int F, int total) {
    int i = blockIdx.x * blockDim.x + threadIdx.x;
    if (i >= total) return;
    int e = i >> gshift;
    int q = (i & ((1 << gshift) - 1)) << 2;
    int s = src[e], d = dst[e];
    float nrm = dinv[s] * dinv[d];
    const float4 hv = *reinterpret_cast<const float4*>(h + (size_t)s * F + q);
    float* a = acc + (size_t)d * F + q;
    atomicAdd(a + 0, nrm * hv.x);
    atomicAdd(a + 1, nrm * hv.y);
    atomicAdd(a + 2, nrm * hv.z);
    atomicAdd(a + 3, nrm * hv.w);
}

// ---------------- LSTM scan (single workgroup; 256KB whh^T staged in LDS) ----------------
// xg[t][j] already contains x-projection + (bih+bhh).  PyTorch gate order i,f,g,o.
__global__ __launch_bounds__(512)
void k_lstm(const float* __restrict__ xg, const float* __restrict__ whh,
            unsigned short* __restrict__ hs_bf, int T) {
    extern __shared__ float sm[];
    float* whh_t = sm;                   // [128][512]  = 256 KB  (CDNA5: 320KB LDS/WGP)
    float* gates = sm + 128 * 512;       // [512]
    float* hh    = gates + 512;          // [128]
    int j = threadIdx.x;                 // 0..511, one gate row per thread

    for (int idx = j; idx < 512 * 128; idx += 512) {   // coalesced read, transposed write
        int r = idx >> 7;                // gate row 0..511
        int k = idx & 127;
        whh_t[k * 512 + r] = whh[idx];
    }
    if (j < 128) hh[j] = 0.0f;
    float cc = 0.0f;
    __syncthreads();

    for (int t = 0; t < T; ++t) {
        float g = xg[(size_t)t * 512 + j];
        const float* wcol = whh_t + j;   // stride-512: lanes hit distinct banks
        #pragma unroll 4
        for (int k = 0; k < 128; ++k) g += wcol[k * 512] * hh[k];
        gates[j] = g;
        __syncthreads();
        if (j < 128) {
            float ig = sigm(gates[j]);
            float fg = sigm(gates[128 + j]);
            float gg = tanhf(gates[256 + j]);
            float og = sigm(gates[384 + j]);
            cc = fg * cc + ig * gg;
            float hv = og * tanhf(cc);
            hh[j] = hv;
            hs_bf[(size_t)t * 128 + j] = f2bf(hv);
        }
        __syncthreads();
    }
}

// ---------------- pooling + head ----------------
__global__ void k_pool(const float* __restrict__ attn, const int* __restrict__ batch,
                       float* __restrict__ sums, float* __restrict__ cnt, int total) {
    int i = blockIdx.x * blockDim.x + threadIdx.x;
    if (i >= total) return;
    int n = i >> 7;             // 128 features
    int f = i & 127;
    int g = batch[n];
    atomicAdd(&sums[g * 128 + f], attn[i]);
    if (f == 0) atomicAdd(&cnt[g], 1.0f);
}

__global__ __launch_bounds__(64)
void k_head(const float* __restrict__ sums, const float* __restrict__ cnt,
            const float* __restrict__ w1, const float* __restrict__ b1,
            const float* __restrict__ w2, const float* __restrict__ b2,
            float* __restrict__ out) {
    __shared__ float pooled[128];
    __shared__ float z1[64];
    __shared__ float logit[2];
    int g = blockIdx.x, j = threadIdx.x;       // 64 threads / graph
    float inv = 1.0f / fmaxf(cnt[g], 1.0f);
    pooled[j]      = sums[g * 128 + j]      * inv;
    pooled[64 + j] = sums[g * 128 + 64 + j] * inv;
    __syncthreads();
    float a = b1[j];
    for (int k = 0; k < 128; ++k) a += pooled[k] * w1[j * 128 + k];
    z1[j] = fmaxf(a, 0.0f);
    __syncthreads();
    if (j < N_CLASSES) {
        float l = b2[j];
        for (int k = 0; k < 64; ++k) l += z1[k] * w2[j * 64 + k];
        logit[j] = l;
    }
    __syncthreads();
    if (j == 0) {
        float m = fmaxf(logit[0], logit[1]);
        float lse = m + __logf(__expf(logit[0] - m) + __expf(logit[1] - m));
        out[g * 2 + 0] = logit[0] - lse;
        out[g * 2 + 1] = logit[1] - lse;
    }
}

// ---------------- launch ----------------
static inline unsigned nblk(long long n, int t) { return (unsigned)((n + t - 1) / t); }

extern "C" void kernel_launch(void* const* d_in, const int* in_sizes, int n_in,
                              void* d_out, int out_size, void* d_ws, size_t ws_size,
                              hipStream_t stream) {
    (void)in_sizes; (void)n_in; (void)out_size; (void)ws_size;
    const float* x     = (const float*)d_in[0];
    const int*   ei    = (const int*)d_in[1];
    const int*   batch = (const int*)d_in[2];
    const float* w1 = (const float*)d_in[3];   const float* b1 = (const float*)d_in[4];
    const float* w2 = (const float*)d_in[5];   const float* b2 = (const float*)d_in[6];
    const float* w3 = (const float*)d_in[7];   const float* b3 = (const float*)d_in[8];
    const float* w4 = (const float*)d_in[9];   const float* b4 = (const float*)d_in[10];
    const float* wih = (const float*)d_in[11]; const float* whh = (const float*)d_in[12];
    const float* bih = (const float*)d_in[13]; const float* bhh = (const float*)d_in[14];
    const float* ain_w  = (const float*)d_in[15]; const float* ain_b  = (const float*)d_in[16];
    const float* aout_w = (const float*)d_in[17]; const float* aout_b = (const float*)d_in[18];
    const float* l1w = (const float*)d_in[19]; const float* l1b = (const float*)d_in[20];
    const float* l2w = (const float*)d_in[21]; const float* l2b = (const float*)d_in[22];
    float* out = (float*)d_out;

    const int* src = ei;
    const int* dst = ei + N_EDGES;

    // ---- workspace carve-out (all buffers rewritten every call) ----
    char* ws = (char*)d_ws;
    size_t off = 0;
    auto carve = [&](size_t bytes) -> char* {
        char* p = ws + off;
        off += (bytes + 255) & ~(size_t)255;
        return p;
    };
    float* deg  = (float*)carve((size_t)N_NODES * 4);
    float* dinv = (float*)carve((size_t)N_NODES * 4);
    float* bufH = (float*)carve((size_t)N_NODES * 512 * 4);   // gemm outputs / xg / attn
    float* bufA = (float*)carve((size_t)N_NODES * 256 * 4);   // GCN aggregation acc
    unsigned short* bufX = (unsigned short*)carve((size_t)N_NODES * 256 * 2); // bf16 acts
    unsigned short* w1p   = (unsigned short*)carve((size_t)128 * 64  * 2);
    unsigned short* w2p   = (unsigned short*)carve((size_t)64  * 64  * 2);
    unsigned short* w3p   = (unsigned short*)carve((size_t)64  * 128 * 2);
    unsigned short* w4p   = (unsigned short*)carve((size_t)128 * 256 * 2);
    unsigned short* wihp  = (unsigned short*)carve((size_t)512 * 256 * 2);
    unsigned short* wvp   = (unsigned short*)carve((size_t)128 * 128 * 2);
    unsigned short* woutp = (unsigned short*)carve((size_t)128 * 128 * 2);
    float* bias_sum = (float*)carve(512 * 4);
    float* sums = (float*)carve((size_t)N_GRAPHS * 128 * 4);
    float* cnt  = (float*)carve((size_t)N_GRAPHS * 4);

    const int T = 256;

    // ---- degree / symmetric norm (self-loops folded as deg init = 1) ----
    k_fill_f32<<<nblk(N_NODES, T), T, 0, stream>>>(deg, 1.0f, N_NODES);
    k_deg<<<nblk(N_EDGES, T), T, 0, stream>>>(dst, deg, N_EDGES);
    k_dinv<<<nblk(N_NODES, T), T, 0, stream>>>(deg, dinv, N_NODES);

    // ---- pack weights to bf16 [Nout][K] ----
    k_pack_w<<<nblk(128 * 64, T),  T, 0, stream>>>(w1, w1p, 128, 64, 1);
    k_pack_w<<<nblk(64 * 64, T),   T, 0, stream>>>(w2, w2p, 64, 64, 1);
    k_pack_w<<<nblk(64 * 128, T),  T, 0, stream>>>(w3, w3p, 64, 128, 1);
    k_pack_w<<<nblk(128 * 256, T), T, 0, stream>>>(w4, w4p, 128, 256, 1);
    k_pack_w<<<nblk(256 * 512, T), T, 0, stream>>>(wih, wihp, 256, 512, 0);
    k_pack_w<<<nblk(128 * 128, T), T, 0, stream>>>(ain_w + 256 * 128, wvp, 128, 128, 0);
    k_pack_w<<<nblk(128 * 128, T), T, 0, stream>>>(aout_w, woutp, 128, 128, 0);
    k_add_vec<<<nblk(512, T), T, 0, stream>>>(bih, bhh, bias_sum, 512);

    // ---- x -> bf16 ----
    k_f32_to_bf16<<<nblk((long long)N_NODES * F_IN, T), T, 0, stream>>>(x, bufX, N_NODES * F_IN, 0);

    // ---- 4 GCN layers ----
    const int Ks[4]     = {128, 64, 64, 128};
    const int Fs[4]     = {64, 64, 128, 256};
    const int fshift[4] = {6, 6, 7, 8};
    const int gshift[4] = {4, 4, 5, 6};
    const float* biases[4] = {b1, b2, b3, b4};
    const unsigned short* wps[4] = {w1p, w2p, w3p, w4p};
    for (int l = 0; l < 4; ++l) {
        int K = Ks[l], F = Fs[l];
        int tiles = (N_NODES / 16) * (F / 16);
        k_gemm_bf16<<<nblk(tiles, 4), 128, 0, stream>>>(bufX, wps[l], nullptr,
                                                        bufH, nullptr, N_NODES, F, K);
        k_gcn_init<<<nblk((long long)N_NODES * F, T), T, 0, stream>>>(bufH, dinv, biases[l],
                                                                      bufA, fshift[l], N_NODES * F);
        int etotal = N_EDGES * (F / 4);
        k_gcn_scatter<<<nblk(etotal, T), T, 0, stream>>>(src, dst, dinv, bufH, bufA,
                                                         gshift[l], F, etotal);
        k_f32_to_bf16<<<nblk((long long)N_NODES * F, T), T, 0, stream>>>(bufA, bufX,
                                                                         N_NODES * F, 1);
    }

    // ---- xg = h @ wih.T + (bih+bhh)  [N,512] ----
    {
        int tiles = (N_NODES / 16) * (512 / 16);
        k_gemm_bf16<<<nblk(tiles, 4), 128, 0, stream>>>(bufX, wihp, bias_sum,
                                                        bufH, nullptr, N_NODES, 512, 256);
    }

    // ---- LSTM scan (serial over N): hs (bf16) -> bufX[0 : N*128) ----
    unsigned short* hsbf = bufX;
    unsigned short* vbf  = bufX + (size_t)N_NODES * 128;
    {
        size_t lds = (size_t)(128 * 512 + 512 + 128) * sizeof(float); // ~258.5 KB
        k_lstm<<<1, 512, lds, stream>>>(bufH, whh, hsbf, N_NODES);
    }

    // ---- attention (L=1 => softmax==1 => ctx = v) ----
    {
        int tiles = (N_NODES / 16) * (128 / 16);
        // v = hs @ Wv.T + bv   (bf16 out only)
        k_gemm_bf16<<<nblk(tiles, 4), 128, 0, stream>>>(hsbf, wvp, ain_b + 256,
                                                        nullptr, vbf, N_NODES, 128, 128);
        // attn = v @ Wout.T + bout  (f32 out in bufH)
        k_gemm_bf16<<<nblk(tiles, 4), 128, 0, stream>>>(vbf, woutp, aout_b,
                                                        bufH, nullptr, N_NODES, 128, 128);
    }

    // ---- global mean pool + MLP head + log_softmax ----
    k_fill_f32<<<nblk(N_GRAPHS * 128, T), T, 0, stream>>>(sums, 0.0f, N_GRAPHS * 128);
    k_fill_f32<<<1, T, 0, stream>>>(cnt, 0.0f, N_GRAPHS);
    k_pool<<<nblk((long long)N_NODES * 128, T), T, 0, stream>>>(bufH, batch, sums, cnt,
                                                                N_NODES * 128);
    k_head<<<N_GRAPHS, 64, 0, stream>>>(sums, cnt, l1w, l1b, l2w, l2b, out);
}
